// SelfAttention_40020505264462
// MI455X (gfx1250) — matrix operand order repaired and checked by reading, not compiled, not run on hardware
//
#include <hip/hip_runtime.h>

// Self-attention (SAGAN-style): B=4, N=H*W=4096, C=256, CK=32, COUT=256.
// bf16 WMMA everywhere (v_wmma_f32_16x16x32_bf16, f32 accum), flash-attention
// streaming over keys. Key/value chunks are staged into LDS by the CDNA5
// Tensor Data Mover (tensor_load_to_lds), double-buffered, shared by the
// 4 waves of each block. B-fragments are preloaded in groups of 8 so each
// K-chunk is one load clause + a WMMA burst instead of 20 wait-gated pairs.

#define BATCH 4
#define NPIX  4096
#define CIN   256
#define CKD   32
#define CO    256

typedef __attribute__((ext_vector_type(16))) __bf16        v16bf;
typedef __attribute__((ext_vector_type(8)))  float         v8f;
typedef __attribute__((ext_vector_type(4)))  unsigned int  v4u;
typedef __attribute__((ext_vector_type(8)))  unsigned int  v8u;

union BF16x16 {
    v16bf v;
    unsigned int u[8];
    uint4 q[2];
};

__device__ __forceinline__ unsigned short f2bf(float f) {
    union { float f; unsigned int u; } c; c.f = f;
    unsigned int r = c.u + 0x7FFFu + ((c.u >> 16) & 1u);
    return (unsigned short)(r >> 16);
}

__device__ __forceinline__ v8f wmma_bf16(const BF16x16& a, const BF16x16& b, v8f c) {
    return __builtin_amdgcn_wmma_f32_16x16x32_bf16(
        false, a.v, false, b.v, (short)0, c, false, false);
}

// A-fragment (16x32 bf16, MxK): lane = row m = lane&15; vgpr j holds the K
// pair k0 = (lane>>4)*16 + j*2 (contiguous, matches load_b_frag).
__device__ __forceinline__ void load_a_frag(BF16x16& A, const unsigned short* base,
                                            int row0, int ld, int kbase, int lane) {
    const int m = lane & 15, h = lane >> 4;
    const unsigned short* p = base + (row0 + m) * ld + kbase;
#pragma unroll
    for (int j = 0; j < 8; ++j) {
        const int k0 = (h << 4) + (j << 1);
        A.u[j] = *(const unsigned int*)(p + k0);
    }
}

// B-fragment (32x16 bf16, KxN): lane = col n = lane&15; 16 contiguous K
// starting at (lane>>4)*16. `base` laid out [N][K] row-major, stride ld.
__device__ __forceinline__ void load_b_frag(BF16x16& Bf, const unsigned short* base,
                                            int col0, int ld, int kbase, int lane) {
    const unsigned short* p = base + (col0 + (lane & 15)) * ld + kbase + ((lane >> 4) << 4);
    Bf.q[0] = *(const uint4*)(p);
    Bf.q[1] = *(const uint4*)(p + 8);
}

// ---------------------------------------------------------------- TDM -------
// 2D tensor_load_to_lds (VADDR2/3 = NULL -> 2-group descriptor).
// Pads the LDS destination 16B per 64B row: 80B LDS row stride, so 16-lane
// B-fragment ds_load_b128 reads hit 16 distinct bank groups (conflict-free).
// Dims/strides in data_size (=2B) units; 64B rows (tile_d0 == 32 elements).
__device__ __forceinline__ void tdm_load_2d(unsigned lds_off, const void* gaddr,
                                            unsigned tensor_d0, unsigned tensor_d1,
                                            unsigned tile_d0, unsigned tile_d1,
                                            unsigned d0_stride) {
    const unsigned long long ga = (unsigned long long)gaddr;
    const v4u g0 = {
        1u,                                               // count=1, user desc
        lds_off,                                          // LDS byte address
        (unsigned)ga,                                     // global_addr[31:0]
        (unsigned)((ga >> 32) & 0x01FFFFFFu) | (2u << 30) // addr[56:32] | type=2
    };
    const v8u g1 = {
        (1u << 16) | (1u << 20) | (3u << 22) | (3u << 25), // 2B elems, pad_en,
                                                           // pad every 16DW by 4DW
        (tensor_d0 & 0xFFFFu) << 16,                       // [63:48] dim0 lo16
        ((tensor_d0 >> 16) & 0xFFFFu) | ((tensor_d1 & 0xFFFFu) << 16),
        ((tensor_d1 >> 16) & 0xFFFFu) | (tile_d0 << 16),   // [127:112] tile_d0
        tile_d1,                                           // [143:128] tile_d1
        d0_stride,                                         // dim0 stride lo32
        0u, 0u
    };
    asm volatile("tensor_load_to_lds %0, %1" :: "s"(g0), "s"(g1) : "memory");
}

__device__ __forceinline__ unsigned lds_off_of(const void* p) {
    return (unsigned)(size_t)p;      // flat LDS pointer: low 32 bits = offset
}

// ---------------------------------------------------------------- prep ------
__global__ void k_cvt_x(const float* __restrict__ x, unsigned short* __restrict__ xb, int n) {
    int i = blockIdx.x * blockDim.x + threadIdx.x;
    if (i < n) xb[i] = f2bf(x[i]);
}

__global__ void k_cvt_wT(const float* __restrict__ w, unsigned short* __restrict__ wt,
                         int rows, int cols) {
    int i = blockIdx.x * blockDim.x + threadIdx.x;
    if (i < rows * cols) {
        int r = i / cols, c = i % cols;
        wt[c * rows + r] = f2bf(w[i]);
    }
}

// ---------------------------------------------------------- projections -----
__global__ __launch_bounds__(128, 1) void k_proj(
    const unsigned short* __restrict__ xb,
    const unsigned short* __restrict__ Wft, const unsigned short* __restrict__ Wgt,
    const unsigned short* __restrict__ Wht,
    const float* __restrict__ bf, const float* __restrict__ bg,
    const float* __restrict__ bh,
    unsigned short* __restrict__ fbuf, unsigned short* __restrict__ gbuf,
    unsigned short* __restrict__ hht)
{
    const int lane = threadIdx.x & 31;
    const int wave = threadIdx.x >> 5;
    const int tile = blockIdx.x * 4 + wave;
    const int b    = tile >> 8;
    const int trow = tile & 255;
    const int rowG = b * NPIX + trow * 16;

    v8f accF[2], accG[2], accH[16];
#pragma unroll
    for (int i = 0; i < 2; ++i) { accF[i] = 0.f; accG[i] = 0.f; }
#pragma unroll
    for (int i = 0; i < 16; ++i) accH[i] = 0.f;

    for (int kc = 0; kc < 8; ++kc) {
        BF16x16 A;
        load_a_frag(A, xb, rowG, CIN, kc * 32, lane);

        // f/g: 4 B-fragments loaded as one clause, then 4 WMMAs
        BF16x16 Bfg[4];
        load_b_frag(Bfg[0], Wft, 0,  CIN, kc * 32, lane);
        load_b_frag(Bfg[1], Wft, 16, CIN, kc * 32, lane);
        load_b_frag(Bfg[2], Wgt, 0,  CIN, kc * 32, lane);
        load_b_frag(Bfg[3], Wgt, 16, CIN, kc * 32, lane);
        accF[0] = wmma_bf16(A, Bfg[0], accF[0]);
        accF[1] = wmma_bf16(A, Bfg[1], accF[1]);
        accG[0] = wmma_bf16(A, Bfg[2], accG[0]);
        accG[1] = wmma_bf16(A, Bfg[3], accG[1]);

        // hh: two groups of 8 fragments -> 16-load clause + 8-WMMA burst
#pragma unroll
        for (int gph = 0; gph < 2; ++gph) {
            BF16x16 Bh8[8];
#pragma unroll
            for (int j = 0; j < 8; ++j)
                load_b_frag(Bh8[j], Wht, (gph * 8 + j) * 16, CIN, kc * 32, lane);
#pragma unroll
            for (int j = 0; j < 8; ++j)
                accH[gph * 8 + j] = wmma_bf16(A, Bh8[j], accH[gph * 8 + j]);
        }
    }

    const int n = lane & 15, h = lane >> 4;
#pragma unroll
    for (int ct = 0; ct < 2; ++ct) {
        const float bfv = bf[ct * 16 + n];
        const float bgv = bg[ct * 16 + n];
#pragma unroll
        for (int r = 0; r < 8; ++r) {
            const int row = rowG + r + 8 * h;
            fbuf[row * CKD + ct * 16 + n] = f2bf(accF[ct][r] + bfv);
            gbuf[row * CKD + ct * 16 + n] = f2bf(accG[ct][r] + bgv);
        }
    }
#pragma unroll
    for (int ct = 0; ct < 16; ++ct) {
        const float bhv = bh[ct * 16 + n];
        union { uint4 q; unsigned short s[8]; } pk;
#pragma unroll
        for (int r = 0; r < 8; ++r) pk.s[r] = f2bf(accH[ct][r] + bhv);
        const int ch = ct * 16 + n;
        *(uint4*)(hht + ((size_t)b * CO + ch) * NPIX + trow * 16 + 8 * h) = pk.q;
    }
}

// ------------------------------------------------------- flash attention ----
// 256 blocks (4 batches x 64 query-blocks), 128 threads = 4 waves.
// Each wave owns a 16-query tile; the block shares one key stream.
// TDM double-buffers the 32-key V tile (16KB) and g tile (2KB) in LDS.
#define LDS_LDV 40   // padded row stride in ushorts (80B = 64B row + 16B pad)

__global__ __launch_bounds__(128, 1) void k_attn(
    const unsigned short* __restrict__ fbuf,
    const unsigned short* __restrict__ gbuf,
    const unsigned short* __restrict__ hht,
    unsigned short* __restrict__ attn)
{
    __shared__ unsigned short ldsV[2][CO * LDS_LDV];   // [ch][key 0..31] padded
    __shared__ unsigned short ldsG[2][32 * LDS_LDV];   // [key][ck 0..31] padded
    __shared__ unsigned short ldsP[4][16 * 32];        // per-wave P transpose

    const int lane = threadIdx.x & 31;
    const int wave = threadIdx.x >> 5;
    const int b    = blockIdx.x >> 6;
    const int qblk = blockIdx.x & 63;
    const int rowG = b * NPIX + qblk * 64 + wave * 16;
    const int n = lane & 15, h = lane >> 4;

    const unsigned short* gB = gbuf + (size_t)b * NPIX * CKD;
    const unsigned short* hB = hht  + (size_t)b * CO * NPIX;

    // preload chunk 0 (wave 0 drives the TDM; EXEC-independent, one issue/wave)
    if (wave == 0) {
        tdm_load_2d(lds_off_of(&ldsV[0][0]), hB, NPIX, CO, 32, CO, NPIX);
        tdm_load_2d(lds_off_of(&ldsG[0][0]), gB, CKD, NPIX, CKD, 32, CKD);
    }

    BF16x16 Af;
    load_a_frag(Af, fbuf, rowG, CKD, 0, lane);

    v8f O[16];
#pragma unroll
    for (int i = 0; i < 16; ++i) O[i] = 0.f;
    float M[8], L[8];
#pragma unroll
    for (int r = 0; r < 8; ++r) { M[r] = -1e30f; L[r] = 0.f; }

    for (int kt = 0; kt < 128; ++kt) {
        const int cur = kt & 1;
        if (wave == 0) {
            if (kt < 127) {   // prefetch chunk kt+1 into the other buffer
                tdm_load_2d(lds_off_of(&ldsV[1 - cur][0]), hB + (kt + 1) * 32,
                            NPIX, CO, 32, CO, NPIX);
                tdm_load_2d(lds_off_of(&ldsG[1 - cur][0]),
                            gB + (size_t)(kt + 1) * 32 * CKD,
                            CKD, NPIX, CKD, 32, CKD);
                // TDM completes in order: <=2 outstanding => chunk kt landed
                __builtin_amdgcn_s_wait_tensorcnt(2);
            } else {
                __builtin_amdgcn_s_wait_tensorcnt(0);
            }
        }
        __syncthreads();                       // buf[cur] ready for all waves

        // S = f @ g^T  (16q x 32keys), K = CK = 32
        BF16x16 Bg0, Bg1;
        v8f S0 = 0.f, S1 = 0.f;
        load_b_frag(Bg0, ldsG[cur], 0,  LDS_LDV, 0, lane);
        load_b_frag(Bg1, ldsG[cur], 16, LDS_LDV, 0, lane);
        S0 = wmma_bf16(Af, Bg0, S0);
        S1 = wmma_bf16(Af, Bg1, S1);

        // online softmax: row max across the 16 lanes of each half
        float scale[8];
#pragma unroll
        for (int r = 0; r < 8; ++r) {
            float cm = fmaxf(S0[r], S1[r]);
#pragma unroll
            for (int msk = 1; msk < 16; msk <<= 1)
                cm = fmaxf(cm, __shfl_xor(cm, msk, 32));
            const float nm = fmaxf(M[r], cm);
            const float sc = __expf(M[r] - nm);
            M[r] = nm;
            const float p0 = __expf(S0[r] - nm);
            const float p1 = __expf(S1[r] - nm);
            L[r] = L[r] * sc + p0 + p1;
            scale[r] = sc;
            const int m = r + 8 * h;
            ldsP[wave][m * 32 + n]      = f2bf(p0);
            ldsP[wave][m * 32 + 16 + n] = f2bf(p1);
        }
#pragma unroll
        for (int ct = 0; ct < 16; ++ct)
#pragma unroll
            for (int r = 0; r < 8; ++r) O[ct][r] *= scale[r];

        // O += P @ V : P as A-fragment via LDS; V B-frags in groups of 8
        BF16x16 Ap;
        load_a_frag(Ap, ldsP[wave], 0, 32, 0, lane);
#pragma unroll
        for (int gph = 0; gph < 2; ++gph) {
            BF16x16 Bh8[8];
#pragma unroll
            for (int j = 0; j < 8; ++j)
                load_b_frag(Bh8[j], ldsV[cur], (gph * 8 + j) * 16, LDS_LDV, 0, lane);
#pragma unroll
            for (int j = 0; j < 8; ++j)
                O[gph * 8 + j] = wmma_bf16(Ap, Bh8[j], O[gph * 8 + j]);
        }
        __syncthreads();                       // all done reading buf[cur]
    }

    // finalize: reduce L across the 16 lanes of each half, normalize, store bf16
    float inv[8];
#pragma unroll
    for (int r = 0; r < 8; ++r) {
        float l = L[r];
#pragma unroll
        for (int msk = 1; msk < 16; msk <<= 1)
            l += __shfl_xor(l, msk, 32);
        inv[r] = 1.0f / l;
    }
#pragma unroll
    for (int ct = 0; ct < 16; ++ct)
#pragma unroll
        for (int r = 0; r < 8; ++r)
            attn[(size_t)(rowG + r + 8 * h) * CO + ct * 16 + n] =
                f2bf(O[ct][r] * inv[r]);
}

// ------------------------------------------------- output proj + residual ---
__global__ __launch_bounds__(128, 1) void k_out(
    const unsigned short* __restrict__ attn,
    const unsigned short* __restrict__ Wot,
    const float* __restrict__ bo, const float* __restrict__ gamma,
    const float* __restrict__ x, float* __restrict__ y)
{
    const int lane = threadIdx.x & 31;
    const int wave = threadIdx.x >> 5;
    const int tile = blockIdx.x * 4 + wave;
    const int b    = tile >> 8;
    const int trow = tile & 255;
    const int rowG = b * NPIX + trow * 16;

    v8f acc[16];
#pragma unroll
    for (int i = 0; i < 16; ++i) acc[i] = 0.f;

    for (int kc = 0; kc < 8; ++kc) {
        BF16x16 A;
        load_a_frag(A, attn, rowG, CO, kc * 32, lane);
#pragma unroll
        for (int gph = 0; gph < 2; ++gph) {
            BF16x16 Bv8[8];
#pragma unroll
            for (int j = 0; j < 8; ++j)
                load_b_frag(Bv8[j], Wot, (gph * 8 + j) * 16, CO, kc * 32, lane);
#pragma unroll
            for (int j = 0; j < 8; ++j)
                acc[gph * 8 + j] = wmma_bf16(A, Bv8[j], acc[gph * 8 + j]);
        }
    }

    const float gm = gamma[0];
    const int n = lane & 15, h = lane >> 4;
#pragma unroll
    for (int ct = 0; ct < 16; ++ct) {
        const float bov = bo[ct * 16 + n];
#pragma unroll
        for (int r = 0; r < 8; ++r) {
            const size_t idx = (size_t)(rowG + r + 8 * h) * CO + ct * 16 + n;
            y[idx] = gm * (acc[ct][r] + bov) + x[idx];
        }
    }
}

// ---------------------------------------------------------------- launch ----
extern "C" void kernel_launch(void* const* d_in, const int* in_sizes, int n_in,
                              void* d_out, int out_size, void* d_ws, size_t ws_size,
                              hipStream_t stream) {
    const float* x  = (const float*)d_in[0];
    const float* Wf = (const float*)d_in[1];
    const float* bf = (const float*)d_in[2];
    const float* Wg = (const float*)d_in[3];
    const float* bg = (const float*)d_in[4];
    const float* Wh = (const float*)d_in[5];
    const float* bh = (const float*)d_in[6];
    const float* Wo = (const float*)d_in[7];
    const float* bo = (const float*)d_in[8];
    const float* gm = (const float*)d_in[9];
    float* y = (float*)d_out;

    char* ws = (char*)d_ws;
    unsigned short* xb  = (unsigned short*)(ws + 0);          //  8 MB  x bf16
    unsigned short* fb  = (unsigned short*)(ws + 8388608);    //  1 MB  f bf16
    unsigned short* gb  = (unsigned short*)(ws + 9437184);    //  1 MB  g bf16
    unsigned short* hht = (unsigned short*)(ws + 10485760);   //  8 MB  h^T bf16
    unsigned short* at  = (unsigned short*)(ws + 18874368);   //  8 MB  attn bf16
    unsigned short* wft = (unsigned short*)(ws + 27262976);   // 16 KB
    unsigned short* wgt = (unsigned short*)(ws + 27279360);   // 16 KB
    unsigned short* wht = (unsigned short*)(ws + 27295744);   // 128 KB
    unsigned short* wot = (unsigned short*)(ws + 27426816);   // 128 KB

    const int nx = BATCH * NPIX * CIN;
    k_cvt_x<<<(nx + 255) / 256, 256, 0, stream>>>(x, xb, nx);
    k_cvt_wT<<<(CIN * CKD + 255) / 256, 256, 0, stream>>>(Wf, wft, CIN, CKD);
    k_cvt_wT<<<(CIN * CKD + 255) / 256, 256, 0, stream>>>(Wg, wgt, CIN, CKD);
    k_cvt_wT<<<(CIN * CO  + 255) / 256, 256, 0, stream>>>(Wh, wht, CIN, CO);
    k_cvt_wT<<<(CO  * CO  + 255) / 256, 256, 0, stream>>>(Wo, wot, CO,  CO);

    k_proj<<<256, 128, 0, stream>>>(xb, wft, wgt, wht, bf, bg, bh, fb, gb, hht);
    k_attn<<<256, 128, 0, stream>>>(fb, gb, hht, at);
    k_out <<<256, 128, 0, stream>>>(at, wot, bo, gm, x, y);
}
